// MultiHeadAttention_33767032881669
// MI455X (gfx1250) — compile-verified, matrix operand order checked by reading, not run on hardware
//
#include <hip/hip_runtime.h>

// ---------------------------------------------------------------------------
// MHA for MI455X (gfx1250): bf16 WMMA pipeline + TDM (tensor_load_to_lds)
//   B=4, S=2048, D=1024, H=16, Dh=64
// ---------------------------------------------------------------------------

#define D_MODEL 1024
#define NUM_HEADS 16
#define HEAD_DIM 64
#define BATCH 4
#define SEQ 2048
#define ROWS (BATCH * SEQ)   // 8192

typedef __attribute__((ext_vector_type(16))) __bf16 v16bf;
typedef __attribute__((ext_vector_type(8)))  __bf16 v8bf;
typedef __attribute__((ext_vector_type(8)))  float  v8f;
typedef __attribute__((ext_vector_type(4)))  unsigned int u32x4;
typedef __attribute__((ext_vector_type(8)))  int i32x8;
typedef __attribute__((ext_vector_type(4)))  int i32x4;

#if defined(__AMDGCN__) && __has_builtin(__builtin_amdgcn_tensor_load_to_lds) && \
    __has_builtin(__builtin_amdgcn_s_wait_tensorcnt)
#define HAVE_TDM 1
#else
#define HAVE_TDM 0
#endif

static __device__ inline v8f wmma_bf16(v16bf a, v16bf b, v8f c) {
    // D = A(16x32 bf16) * B(32x16 bf16) + C(16x16 f32)
    return __builtin_amdgcn_wmma_f32_16x16x32_bf16(
        false, a, false, b, (short)0, c, false, false);
}

// One A/B fragment: 16 bf16 per lane as two 16-byte chunks at +0 / +16 elems
// (ISA 16-bit matrix layout: lane row, k offsets {half*8..} and {16+half*8..}).
static __device__ inline v16bf load_frag(const __bf16* p) {
    v8bf lo = *reinterpret_cast<const v8bf*>(p);
    v8bf hi = *reinterpret_cast<const v8bf*>(p + 16);
    v16bf r;
#pragma unroll
    for (int i = 0; i < 8; ++i) { r[i] = lo[i]; r[i + 8] = hi[i]; }
    return r;
}

#if HAVE_TDM
// 2D TDM load: rows x d0_elems (bf16) from global into LDS at lds_byte_off,
// with LDS row padding so padded row stride stays 16B-aligned & conflict-free.
// pad_int: interval encoding (3 = every 16 DWORDs, 4 = every 32 DWORDs)
// pad_amt: amount encoding  (3 = 4 DWORDs = 16 bytes = 8 bf16)
static __device__ inline void tdm_load_2d(unsigned lds_byte_off, const void* gptr,
                                          unsigned d0_elems, unsigned rows,
                                          unsigned stride_elems,
                                          unsigned pad_int, unsigned pad_amt) {
    unsigned long long ga = (unsigned long long)(size_t)gptr;
    u32x4 g0 = {};
    g0[0] = 1u;                                              // count=1 (valid D#)
    g0[1] = lds_byte_off;                                    // lds_addr
    g0[2] = (unsigned)ga;                                    // global_addr[31:0]
    g0[3] = (unsigned)((ga >> 32) & 0x01ffffffu) | 0x80000000u; // [56:32] | type=2
    i32x8 g1 = {};
    g1[0] = (int)((1u << 16) |            // data_size = 2 bytes
                  (1u << 20) |            // pad_enable
                  (pad_int << 22) | (pad_amt << 25));
    g1[1] = (int)((d0_elems & 0xffffu) << 16);               // tensor_dim0 lo
    g1[2] = (int)((d0_elems >> 16) | ((rows & 0xffffu) << 16)); // dim0 hi | dim1 lo
    g1[3] = (int)((rows >> 16) | ((d0_elems & 0xffffu) << 16)); // dim1 hi | tile_dim0
    g1[4] = (int)(rows & 0xffffu);                           // tile_dim1 (tile_dim2=0)
    g1[5] = (int)stride_elems;                               // tensor_dim0_stride lo32
    i32x4 z4 = {};
#if __clang_major__ >= 23
    i32x8 z8 = {};
    __builtin_amdgcn_tensor_load_to_lds(g0, g1, z4, z4, z8, 0);
#else
    __builtin_amdgcn_tensor_load_to_lds(g0, g1, z4, z4, 0);
#endif
}
#endif

// ---------------------------------------------------------------------------
// fp32 -> bf16 bulk convert (vectorized x4)
// ---------------------------------------------------------------------------
__global__ void cvt_f32_bf16_kernel(const float* __restrict__ in,
                                    __bf16* __restrict__ out, int n4) {
    int i = blockIdx.x * blockDim.x + threadIdx.x;
    if (i < n4) {
        float4 v = reinterpret_cast<const float4*>(in)[i];
        int o = i * 4;
        out[o + 0] = (__bf16)v.x;
        out[o + 1] = (__bf16)v.y;
        out[o + 2] = (__bf16)v.z;
        out[o + 3] = (__bf16)v.w;
    }
}

// ---------------------------------------------------------------------------
// W [K=1024][N=1024] fp32  ->  Wt [N][K] bf16   (LDS tile transpose)
// ---------------------------------------------------------------------------
__global__ __launch_bounds__(256) void transpose_w_kernel(
    const float* __restrict__ W, __bf16* __restrict__ Wt) {
    __shared__ float tile[32][33];
    int tx = threadIdx.x & 31;
    int ty = threadIdx.x >> 5;                 // 0..7
    int nb = blockIdx.x * 32;
    int kb = blockIdx.y * 32;
#pragma unroll
    for (int i = 0; i < 32; i += 8)
        tile[ty + i][tx] = W[(size_t)(kb + ty + i) * D_MODEL + nb + tx];
    __syncthreads();
#pragma unroll
    for (int i = 0; i < 32; i += 8)
        Wt[(size_t)(nb + ty + i) * D_MODEL + kb + tx] = (__bf16)tile[tx][ty + i];
}

// ---------------------------------------------------------------------------
// GEMM: out = A[ROWS,1024](bf16) * W (Wt pre-transposed [N,K] bf16) + bias
//   mode 0: fp32 out flat [ROWS,1024]   mode 1: bf16 [B,H,S,64]
//   mode 3: bf16 [B,H,64,S]
// Block 128 threads (4 waves), block tile 128x64, wave tile 32x64.
// Weight tile (64 cols x 32 k) is shared by all 4 waves -> staged in LDS via
// TDM, double buffered (2 x 64 rows x 40-elem padded rows = 10240 B).
// ---------------------------------------------------------------------------
__global__ __launch_bounds__(128) void gemm_bf16_kernel(
    const __bf16* __restrict__ A, const __bf16* __restrict__ Wt,
    const float* __restrict__ bias, void* __restrict__ out, int mode) {
    extern __shared__ __align__(16) char smem_raw[];
    __bf16* sm = (__bf16*)smem_raw;         // 2 buffers x 64*40 elems
    const int K = D_MODEL;
    const int wave = threadIdx.x >> 5;
    const int lane = threadIdx.x & 31;
    const int m16 = lane & 15;
    const int half = lane >> 4;
    const int c0 = blockIdx.x * 64;
    const int r0 = blockIdx.y * 128 + wave * 32;

    v8f acc[2][4] = {};
    const __bf16* arow0 = A + (size_t)(r0 + m16) * K + half * 8;
    const __bf16* arow1 = A + (size_t)(r0 + 16 + m16) * K + half * 8;

#if HAVE_TDM
    if (wave == 0)
        tdm_load_2d(0u, Wt + (size_t)c0 * K, 32u, 64u, (unsigned)K, 3u, 3u);
    int buf = 0;
    for (int kb = 0; kb < K; kb += 32) {
        const bool more = (kb + 32) < K;
        if (wave == 0) {
            if (more) {
                tdm_load_2d((unsigned)((buf ^ 1) * 64 * 40 * 2),
                            Wt + (size_t)c0 * K + (kb + 32),
                            32u, 64u, (unsigned)K, 3u, 3u);
                __builtin_amdgcn_s_wait_tensorcnt(1);
            } else {
                __builtin_amdgcn_s_wait_tensorcnt(0);
            }
        }
        __syncthreads();
        v16bf a0 = load_frag(arow0 + kb);
        v16bf a1 = load_frag(arow1 + kb);
        const __bf16* bt = sm + buf * (64 * 40);
        v16bf bfr[4];
#pragma unroll
        for (int f = 0; f < 4; ++f)
            bfr[f] = load_frag(bt + (f * 16 + m16) * 40 + half * 8);
#pragma unroll
        for (int f = 0; f < 4; ++f) {
            acc[0][f] = wmma_bf16(a0, bfr[f], acc[0][f]);
            acc[1][f] = wmma_bf16(a1, bfr[f], acc[1][f]);
        }
        __syncthreads();
        buf ^= 1;
    }
#else
    const __bf16* brow[4];
#pragma unroll
    for (int f = 0; f < 4; ++f)
        brow[f] = Wt + (size_t)(c0 + f * 16 + m16) * K + half * 8;
    for (int kb = 0; kb < K; kb += 32) {
        __builtin_prefetch(arow0 + kb + 128, 0, 3);
        v16bf a0 = load_frag(arow0 + kb);
        v16bf a1 = load_frag(arow1 + kb);
        v16bf bfr[4];
#pragma unroll
        for (int f = 0; f < 4; ++f) bfr[f] = load_frag(brow[f] + kb);
#pragma unroll
        for (int f = 0; f < 4; ++f) {
            acc[0][f] = wmma_bf16(a0, bfr[f], acc[0][f]);
            acc[1][f] = wmma_bf16(a1, bfr[f], acc[1][f]);
        }
    }
#endif

#pragma unroll
    for (int rf = 0; rf < 2; ++rf) {
#pragma unroll
        for (int f = 0; f < 4; ++f) {
            const int col = c0 + f * 16 + m16;
            const float bv = bias[col];
#pragma unroll
            for (int r = 0; r < 8; ++r) {
                const int row = r0 + rf * 16 + r + 8 * half;
                const float v = acc[rf][f][r] + bv;
                if (mode == 0) {
                    ((float*)out)[(size_t)row * D_MODEL + col] = v;
                } else {
                    const int b = row >> 11;
                    const int s = row & (SEQ - 1);
                    const int h = col >> 6;
                    const int dh = col & 63;
                    size_t idx;
                    if (mode == 3)   // V transposed: [B,H,64,S]
                        idx = ((size_t)(b * NUM_HEADS + h) * HEAD_DIM + dh) * SEQ + s;
                    else             // Q/K: [B,H,S,64]
                        idx = ((size_t)(b * NUM_HEADS + h) * SEQ + s) * HEAD_DIM + dh;
                    ((__bf16*)out)[idx] = (__bf16)v;
                }
            }
        }
    }
}

// ---------------------------------------------------------------------------
// Flash attention: block = 4 waves, 64 queries (4 q-tiles) of one (b,h).
// Per 32-key step the shared K tile (32x64) and V tile (64x32, from Vt) are
// staged in LDS via TDM (double buffered); every wave builds fragments with
// ds_load_b128. P relayout (C-frag -> A-frag) goes through a per-wave LDS
// scratch (same-wave DS ops are in order).
// LDS (elems): K: 2x32x72  V: 2x64x40  P: 4x16x40  -> 24576 bytes.
// ---------------------------------------------------------------------------
#define K_ELE (32 * 72)
#define V_ELE (64 * 40)
#define ATTN_LDS_BYTES ((2 * K_ELE + 2 * V_ELE + 4 * 16 * 40) * 2)

__global__ __launch_bounds__(128) void attn_kernel(
    const __bf16* __restrict__ Q, const __bf16* __restrict__ Kb,
    const __bf16* __restrict__ Vt, __bf16* __restrict__ AO) {
    extern __shared__ __align__(16) char smem_raw[];
    __bf16* sm = (__bf16*)smem_raw;
    const int bid = blockIdx.x;             // b*H*(S/64) + h*(S/64) + qblk
    const int qblk = bid & 31;              // S/64 = 32
    const int h = (bid >> 5) & (NUM_HEADS - 1);
    const int b = bid >> 9;
    const int wave = threadIdx.x >> 5;
    const int lane = threadIdx.x & 31;
    const int m16 = lane & 15;
    const int half = lane >> 4;
    const int q0 = qblk * 64 + wave * 16;

    __bf16* pbuf = sm + (2 * K_ELE + 2 * V_ELE) + wave * (16 * 40);

    const size_t bh = (size_t)(b * NUM_HEADS + h);
    const __bf16* qp = Q + (bh * SEQ + q0 + m16) * HEAD_DIM + half * 8;
    const v16bf qa0 = load_frag(qp);          // d = 0..31
    const v16bf qa1 = load_frag(qp + 32);     // d = 32..63

    const __bf16* kbase = Kb + bh * SEQ * HEAD_DIM;
    const __bf16* vbase = Vt + bh * HEAD_DIM * SEQ;

    float mrow[8], lrow[8];
#pragma unroll
    for (int r = 0; r < 8; ++r) { mrow[r] = -1e30f; lrow[r] = 0.0f; }
    v8f o[4] = {};
    const float scale = 0.125f;  // 1/sqrt(64)

#if HAVE_TDM
    if (wave == 0) {
        tdm_load_2d(0u, kbase, 64u, 32u, 64u, 4u, 3u);
        tdm_load_2d((unsigned)(2 * K_ELE * 2), vbase, 32u, 64u, (unsigned)SEQ, 3u, 3u);
    }
    int buf = 0;
#endif

    for (int kt = 0; kt < SEQ; kt += 32) {
#if HAVE_TDM
        const bool more = (kt + 32) < SEQ;
        if (wave == 0) {
            if (more) {
                tdm_load_2d((unsigned)((buf ^ 1) * K_ELE * 2),
                            kbase + (size_t)(kt + 32) * HEAD_DIM,
                            64u, 32u, 64u, 4u, 3u);
                tdm_load_2d((unsigned)((2 * K_ELE + (buf ^ 1) * V_ELE) * 2),
                            vbase + (kt + 32), 32u, 64u, (unsigned)SEQ, 3u, 3u);
                __builtin_amdgcn_s_wait_tensorcnt(2);
            } else {
                __builtin_amdgcn_s_wait_tensorcnt(0);
            }
        }
        __syncthreads();
        const __bf16* kt_lds = sm + buf * K_ELE;
        const __bf16* vt_lds = sm + 2 * K_ELE + buf * V_ELE;
#endif

        // ---- scores S = Q * K^T for 32 keys (two 16-key C fragments) ----
        v8f s0 = {}, s1 = {};
        {
#if HAVE_TDM
            v16bf kb00 = load_frag(kt_lds + (0 + m16) * 72 + 0 + half * 8);
            v16bf kb01 = load_frag(kt_lds + (0 + m16) * 72 + 32 + half * 8);
            v16bf kb10 = load_frag(kt_lds + (16 + m16) * 72 + 0 + half * 8);
            v16bf kb11 = load_frag(kt_lds + (16 + m16) * 72 + 32 + half * 8);
#else
            const __bf16* k0p = kbase + (size_t)(kt + m16) * HEAD_DIM + half * 8;
            const __bf16* k1p = kbase + (size_t)(kt + 16 + m16) * HEAD_DIM + half * 8;
            v16bf kb00 = load_frag(k0p);
            v16bf kb01 = load_frag(k0p + 32);
            v16bf kb10 = load_frag(k1p);
            v16bf kb11 = load_frag(k1p + 32);
#endif
            s0 = wmma_bf16(qa0, kb00, s0);
            s0 = wmma_bf16(qa1, kb01, s0);
            s1 = wmma_bf16(qa0, kb10, s1);
            s1 = wmma_bf16(qa1, kb11, s1);
        }

        // ---- online softmax over this 32-key chunk ----
        float p0[8], p1[8], tmax[8], tsum[8];
#pragma unroll
        for (int r = 0; r < 8; ++r) {
            p0[r] = s0[r] * scale;
            p1[r] = s1[r] * scale;
            tmax[r] = fmaxf(p0[r], p1[r]);
        }
#pragma unroll
        for (int off = 1; off < 16; off <<= 1) {
#pragma unroll
            for (int r = 0; r < 8; ++r)
                tmax[r] = fmaxf(tmax[r], __shfl_xor(tmax[r], off, 32));
        }
        float corr[8];
#pragma unroll
        for (int r = 0; r < 8; ++r) {
            const float nm = fmaxf(mrow[r], tmax[r]);
            corr[r] = __expf(mrow[r] - nm);
            p0[r] = __expf(p0[r] - nm);
            p1[r] = __expf(p1[r] - nm);
            tsum[r] = p0[r] + p1[r];
            mrow[r] = nm;
        }
#pragma unroll
        for (int off = 1; off < 16; off <<= 1) {
#pragma unroll
            for (int r = 0; r < 8; ++r)
                tsum[r] += __shfl_xor(tsum[r], off, 32);
        }
#pragma unroll
        for (int r = 0; r < 8; ++r) lrow[r] = lrow[r] * corr[r] + tsum[r];
#pragma unroll
        for (int f = 0; f < 4; ++f)
#pragma unroll
            for (int r = 0; r < 8; ++r) o[f][r] *= corr[r];

        // ---- relayout P (C fragment) -> A fragment through per-wave LDS ----
#pragma unroll
        for (int r = 0; r < 8; ++r) {
            const int row = r + 8 * half;
            pbuf[row * 40 + m16] = (__bf16)p0[r];
            pbuf[row * 40 + 16 + m16] = (__bf16)p1[r];
        }
        const v16bf pa = load_frag(pbuf + m16 * 40 + half * 8);

        // ---- O += P * V ----
#pragma unroll
        for (int f = 0; f < 4; ++f) {
#if HAVE_TDM
            v16bf vb = load_frag(vt_lds + (f * 16 + m16) * 40 + half * 8);
#else
            v16bf vb = load_frag(vbase + (size_t)(f * 16 + m16) * SEQ + kt + half * 8);
#endif
            o[f] = wmma_bf16(pa, vb, o[f]);
        }

#if HAVE_TDM
        __syncthreads();
        buf ^= 1;
#endif
    }

    // ---- normalize and write to [ROWS, D_MODEL] bf16 ----
    float inv[8];
#pragma unroll
    for (int r = 0; r < 8; ++r) inv[r] = 1.0f / lrow[r];
#pragma unroll
    for (int f = 0; f < 4; ++f) {
        const int col = h * HEAD_DIM + f * 16 + m16;
#pragma unroll
        for (int r = 0; r < 8; ++r) {
            const int row = q0 + r + 8 * half;
            AO[((size_t)b * SEQ + row) * D_MODEL + col] = (__bf16)(o[f][r] * inv[r]);
        }
    }
}

// ---------------------------------------------------------------------------
// Launcher
// ---------------------------------------------------------------------------
extern "C" void kernel_launch(void* const* d_in, const int* in_sizes, int n_in,
                              void* d_out, int out_size, void* d_ws, size_t ws_size,
                              hipStream_t stream) {
    (void)in_sizes; (void)n_in; (void)out_size; (void)ws_size;

    const float* x  = (const float*)d_in[0];
    const float* Wq = (const float*)d_in[1];
    const float* bq = (const float*)d_in[2];
    const float* Wk = (const float*)d_in[3];
    const float* bk = (const float*)d_in[4];
    const float* Wv = (const float*)d_in[5];
    const float* bv = (const float*)d_in[6];
    const float* Wo = (const float*)d_in[7];
    const float* bo = (const float*)d_in[8];

    char* ws = (char*)d_ws;
    const size_t SZ_X = (size_t)ROWS * D_MODEL * sizeof(__bf16);      // 16 MB
    const size_t SZ_W = (size_t)D_MODEL * D_MODEL * sizeof(__bf16);   // 2 MB
    __bf16* xb  = (__bf16*)ws;                    ws += SZ_X;
    __bf16* Wqt = (__bf16*)ws;                    ws += SZ_W;
    __bf16* Wkt = (__bf16*)ws;                    ws += SZ_W;
    __bf16* Wvt = (__bf16*)ws;                    ws += SZ_W;
    __bf16* Wot = (__bf16*)ws;                    ws += SZ_W;
    __bf16* Qb  = (__bf16*)ws;                    ws += SZ_X;
    __bf16* Kbf = (__bf16*)ws;                    ws += SZ_X;
    __bf16* Vtb = (__bf16*)ws;                    ws += SZ_X;
    __bf16* AO  = (__bf16*)ws;                    ws += SZ_X;

    const size_t GEMM_LDS = 2 * 64 * 40 * sizeof(__bf16);   // 10240 B

    // 1) precision conversion + weight transpose
    {
        const int n4 = ROWS * D_MODEL / 4;
        cvt_f32_bf16_kernel<<<(n4 + 255) / 256, 256, 0, stream>>>(x, xb, n4);
        dim3 g(32, 32);
        transpose_w_kernel<<<g, 256, 0, stream>>>(Wq, Wqt);
        transpose_w_kernel<<<g, 256, 0, stream>>>(Wk, Wkt);
        transpose_w_kernel<<<g, 256, 0, stream>>>(Wv, Wvt);
        transpose_w_kernel<<<g, 256, 0, stream>>>(Wo, Wot);
    }

    // 2) Q/K/V projections
    {
        dim3 grid(D_MODEL / 64, ROWS / 128);   // 16 x 64
        gemm_bf16_kernel<<<grid, 128, GEMM_LDS, stream>>>(xb, Wqt, bq, (void*)Qb, 1);
        gemm_bf16_kernel<<<grid, 128, GEMM_LDS, stream>>>(xb, Wkt, bk, (void*)Kbf, 1);
        gemm_bf16_kernel<<<grid, 128, GEMM_LDS, stream>>>(xb, Wvt, bv, (void*)Vtb, 3);
    }

    // 3) attention: 4 waves/block, 64 queries/block
    attn_kernel<<<BATCH * NUM_HEADS * (SEQ / 64), 128, ATTN_LDS_BYTES, stream>>>(
        Qb, Kbf, Vtb, AO);

    // 4) output projection -> fp32 d_out
    {
        dim3 grid(D_MODEL / 64, ROWS / 128);
        gemm_bf16_kernel<<<grid, 128, GEMM_LDS, stream>>>(AO, Wot, bo, d_out, 0);
    }
}
